// QuantizedLinear_29463475650732
// MI455X (gfx1250) — matrix operand order repaired and checked by reading, not compiled, not run on hardware
//
#include <hip/hip_runtime.h>

typedef __attribute__((ext_vector_type(16))) _Float16 v16h;
typedef __attribute__((ext_vector_type(8)))  _Float16 v8h;
typedef __attribute__((ext_vector_type(2)))  _Float16 h2;
typedef __attribute__((ext_vector_type(8)))  float    v8f;
typedef __attribute__((ext_vector_type(4)))  float    f32x4;
typedef __attribute__((ext_vector_type(4)))  int      i32x4;

#define BM 128
#define BN 128
#define BK 64
#define LDK (BK + 8)   // +8 halfwords padding -> 144B row stride, spreads LDS banks
#define NT  512        // 16 waves

__global__ __launch_bounds__(NT) void qlinear_wmma_f16(
    const float* __restrict__ x,      // [M, K] fp32
    const int*   __restrict__ qw,     // [N, K/2] one packed byte per int32
    const float* __restrict__ scale,  // [N, K/128]
    const float* __restrict__ zpt,    // [N, K/128]
    float*       __restrict__ out,    // [M, N] fp32
    int M, int K, int N)
{
  __shared__ _Float16 As[BM][LDK];   // A tile: x converted to f16
  __shared__ _Float16 Bs[BN][LDK];   // B tile: dequantized weights f16 (row = n, col = k)

  const int t  = threadIdx.x;
  const int m0 = blockIdx.y * BM;
  const int n0 = blockIdx.x * BN;

  // ---- global staging assignment (4 threads per tile row) ----
  const int arow = t >> 2;             // 0..127 : row of A tile
  const int acol = (t & 3) * 16;       // f32 k-offset within BK (16 floats each)
  const int brow = t >> 2;             // 0..127 : row of B tile (= output col)
  const int bj   = (t & 3) * 8;        // packed-int32 offset within 32 per row

  const float* xg = x  + (size_t)(m0 + arow) * K;
  const int*   qg = qw + (size_t)(n0 + brow) * (K / 2);
  const int    ngrp = K / 128;
  const int    kblocks = K / BK;

  f32x4 abuf[4];   // 16 floats of x
  i32x4 bbuf[2];   // 8 packed bytes -> 16 weight nibbles
  float sbuf, zbuf;

  auto stage = [&](int kb) {
    const f32x4* xp = (const f32x4*)(xg + kb * BK + acol);
#pragma unroll
    for (int i = 0; i < 4; ++i) abuf[i] = xp[i];
    const i32x4* qp = (const i32x4*)(qg + kb * (BK / 2) + bj);
#pragma unroll
    for (int i = 0; i < 2; ++i) bbuf[i] = qp[i];
    const int g = (kb * BK) / 128;     // one quant group per K-block (BK=64 | 128)
    sbuf = scale[(size_t)(n0 + brow) * ngrp + g];
    zbuf = zpt  [(size_t)(n0 + brow) * ngrp + g];
  };

  auto prefetch2 = [&](int kb) {       // L2 prefetch for the block after the staged one
    __builtin_prefetch(xg + kb * BK + acol, 0, 1);
    __builtin_prefetch(qg + kb * (BK / 2) + bj, 0, 1);
  };

  auto commit = [&]() {
    _Float16* ap = &As[arow][acol];
#pragma unroll
    for (int i = 0; i < 4; ++i) {
      ap[i * 4 + 0] = (_Float16)abuf[i].x;
      ap[i * 4 + 1] = (_Float16)abuf[i].y;
      ap[i * 4 + 2] = (_Float16)abuf[i].z;
      ap[i * 4 + 3] = (_Float16)abuf[i].w;
    }
    // f16 bit-trick dequant: nibble n -> f16(1024+n) via exponent-bias OR, then
    // one packed FMA: (1024+n)*s - (1024+zp)*s = (n-zp)*s.  c is exact for zp=0.
    const _Float16 sh  = (_Float16)sbuf;
    const float    shf = (float)sh;
    const _Float16 chh = (_Float16)(-(1024.0f + zbuf) * shf);
    const h2 s2 = {sh, sh};
    const h2 c2 = {chh, chh};
    h2* bp = (h2*)&Bs[brow][bj * 2];
#pragma unroll
    for (int i = 0; i < 2; ++i) {
#pragma unroll
      for (int j = 0; j < 4; ++j) {
        const unsigned u = (unsigned)bbuf[i][j];
        const unsigned p = (((u << 12) | u) & 0x000F000Fu) | 0x64006400u;
        const h2 val = __builtin_bit_cast(h2, p);
        bp[i * 4 + j] = val * s2 + c2;   // v_pk_fma_f16
      }
    }
  };

  // ---- wave tiling: 16 waves as 4(M) x 4(N); each wave 32x32 = 2x2 WMMA tiles ----
  const int wave = t >> 5;
  const int lane = t & 31;
  const int hi   = lane >> 4;          // half-wave select
  const int l16  = lane & 15;
  const int wm   = (wave >> 2) * 32;   // 0,32,64,96
  const int wn   = (wave & 3) * 32;    // 0,32,64,96

  v8f acc[2][2];
#pragma unroll
  for (int mf = 0; mf < 2; ++mf)
#pragma unroll
    for (int nf = 0; nf < 2; ++nf)
      acc[mf][nf] = (v8f){0.f, 0.f, 0.f, 0.f, 0.f, 0.f, 0.f, 0.f};

  stage(0);
  for (int kb = 0; kb < kblocks; ++kb) {
    commit();
    __syncthreads();
    if (kb + 1 < kblocks) stage(kb + 1);     // overlap next global loads with WMMAs
    if (kb + 2 < kblocks) prefetch2(kb + 2); // global_prefetch_b8 toward L2

#pragma unroll
    for (int ks = 0; ks < 2; ++ks) {         // two 16x16x32 K-steps per BK=64
      const int kbase = ks * 32;

      v16h afrag[2];
#pragma unroll
      for (int mf = 0; mf < 2; ++mf) {
        // A 16x32 f16 layout: lanes0-15 K0-7 then K16-23; lanes16-31 K8-15 then K24-31
        const v8h* p0 = (const v8h*)&As[wm + mf * 16 + l16][kbase + hi * 8];
        const v8h* p1 = (const v8h*)&As[wm + mf * 16 + l16][kbase + 16 + hi * 8];
        ((v8h*)&afrag[mf])[0] = *p0;
        ((v8h*)&afrag[mf])[1] = *p1;
      }

      v16h bfrag[2];
#pragma unroll
      for (int nf = 0; nf < 2; ++nf) {
        // B 32x16 f16 layout: lane n holds column n; lanes0-15 K0-7 then K16-23; lanes16-31 K8-15 then K24-31
        const v8h* q0 = (const v8h*)&Bs[wn + nf * 16 + l16][kbase + hi * 8];
        const v8h* q1 = (const v8h*)&Bs[wn + nf * 16 + l16][kbase + 16 + hi * 8];
        ((v8h*)&bfrag[nf])[0] = *q0;
        ((v8h*)&bfrag[nf])[1] = *q1;
      }

#pragma unroll
      for (int mf = 0; mf < 2; ++mf)
#pragma unroll
        for (int nf = 0; nf < 2; ++nf)
          acc[mf][nf] = __builtin_amdgcn_wmma_f32_16x16x32_f16(
              false, afrag[mf], false, bfrag[nf],
              (short)0, acc[mf][nf], false, false);
    }
    __syncthreads();
  }

  // ---- store: C layout VGPR r -> (M = r + hi*8, N = l16) ----
#pragma unroll
  for (int mf = 0; mf < 2; ++mf) {
#pragma unroll
    for (int nf = 0; nf < 2; ++nf) {
      const int n = n0 + wn + nf * 16 + l16;
#pragma unroll
      for (int r = 0; r < 8; ++r) {
        const int m = m0 + wm + mf * 16 + hi * 8 + r;
        out[(size_t)m * N + n] = acc[mf][nf][r];
      }
    }
  }
}

extern "C" void kernel_launch(void* const* d_in, const int* in_sizes, int n_in,
                              void* d_out, int out_size, void* d_ws, size_t ws_size,
                              hipStream_t stream) {
  const float* x  = (const float*)d_in[0];
  const int*   qw = (const int*)d_in[1];
  const float* sc = (const float*)d_in[2];
  const float* zp = (const float*)d_in[3];
  float*       y  = (float*)d_out;

  const int K   = 4096;                        // IN (reference)
  const int M   = in_sizes[0] / K;             // B*S = 8192
  const int N   = (in_sizes[1] * 2) / K;       // OUT = 11008

  dim3 grid(N / BN, M / BM);                   // 86 x 64
  qlinear_wmma_f16<<<grid, NT, 0, stream>>>(x, qw, sc, zp, y, M, K, N);
}
